// MultiHeadAttention_28570122453104
// MI455X (gfx1250) — compile-verified, matrix operand order checked by reading, not run on hardware
//
#include <hip/hip_runtime.h>
#include <hip/hip_bf16.h>

// ---------------- problem constants ----------------
#define BATCH  8
#define HEADS  16
#define SEQ    1024
#define DMODEL 1024
#define DH     64   // DMODEL / HEADS

typedef __attribute__((ext_vector_type(16))) _Float16 v16h;
typedef __attribute__((ext_vector_type(8)))  _Float16 v8h;
typedef __attribute__((ext_vector_type(4)))  _Float16 v4h;
typedef __attribute__((ext_vector_type(8)))  float    v8f;
typedef __attribute__((ext_vector_type(4)))  float    v4f;

// ---------------- WMMA fragment helpers ----------------
// 16-bit A fragment 16x32 (and B fragment 32x16, which mirrors it):
//   element h of the v16h maps to K = (h<8?0:16) + hiSel + (h&7),
//   hiSel = 8 for lanes 16..31, 0 for lanes 0..15.
// rowBase points at K=0 of the 32-wide chunk for this lane's row (A) / col (B),
// with the K dimension contiguous in memory. Both 8-half runs are 16B aligned.
__device__ __forceinline__ v16h load_frag16(const _Float16* rowBase, int hiSel) {
  v8h lo = *(const v8h*)(rowBase + hiSel);        // K = hiSel + 0..7
  v8h hi = *(const v8h*)(rowBase + 16 + hiSel);   // K = 16 + hiSel + 0..7
  v16h r;
#pragma unroll
  for (int i = 0; i < 8; ++i) { r[i] = lo[i]; r[i + 8] = hi[i]; }
  return r;
}

__device__ __forceinline__ v8f wmma_f16(v16h a, v16h b, v8f c) {
  return __builtin_amdgcn_wmma_f32_16x16x32_f16(false, a, false, b,
                                                (short)0, c, false, false);
}

// ---------------- LDS layout for attn_kernel (dynamic LDS) ----------------
constexpr int SMEM_SCORES = 0;                          // float [16][1024]  64 KB
constexpr int SMEM_P      = 16 * SEQ * 4;               // f16   [16][1024]  32 KB
constexpr int SMEM_RED    = SMEM_P + 16 * SEQ * 2;      // float [16][8]
constexpr int SMEM_STAT   = SMEM_RED + 16 * 8 * 4;      // float [16]
constexpr int SMEM_CTX    = SMEM_STAT + 16 * 4;         // float [16][64]     4 KB
constexpr int SMEM_TOTAL  = SMEM_CTX + 16 * DH * 4;     // ~101 KB (< 320 KB WGP LDS)

// ---------------- kernel 1: f32 -> f16 relayout of Q,K,V ----------------
__global__ void __launch_bounds__(256)
convert_qkv(const float* __restrict__ q, const float* __restrict__ k,
            const float* __restrict__ v, _Float16* __restrict__ Qh,
            _Float16* __restrict__ Kh, _Float16* __restrict__ Vt) {
  size_t idx = (size_t)blockIdx.x * 256 + threadIdx.x;
  if (idx >= (size_t)BATCH * SEQ * DMODEL) return;
  int d = (int)(idx % DMODEL);
  size_t t = idx / DMODEL;
  int s = (int)(t % SEQ);
  int b = (int)(t / SEQ);
  int head = d >> 6, dd = d & 63;
  size_t hm = ((size_t)(b * HEADS + head) * SEQ + s) * DH + dd;      // [b,h,s,dh]
  Qh[hm] = (_Float16)(q[idx] * 0.125f);   // fold 1/sqrt(dh) (exact pow2) into Q
  Kh[hm] = (_Float16)k[idx];
  Vt[((size_t)(b * HEADS + head) * DH + dd) * SEQ + s] = (_Float16)v[idx]; // [b,h,dh,s]
}

// ---------------- kernel 2: W f32 -> f16 ----------------
__global__ void __launch_bounds__(256)
convert_w(const float* __restrict__ w, _Float16* __restrict__ Wh) {
  int idx = blockIdx.x * 256 + threadIdx.x;
  if (idx < DMODEL * DMODEL) Wh[idx] = (_Float16)w[idx];
}

// ---------------- kernel 3: attention (scores + softmax + P@V) ----------------
// one workgroup = 4 waves = one (b, h, 16-row query tile)
__global__ void __launch_bounds__(128)
attn_kernel(const _Float16* __restrict__ Qh, const _Float16* __restrict__ Kh,
            const _Float16* __restrict__ Vt, float* __restrict__ attn,
            _Float16* __restrict__ Ctx) {
  extern __shared__ char smem[];
  float*    s_scores = (float*)(smem + SMEM_SCORES);
  _Float16* s_p      = (_Float16*)(smem + SMEM_P);
  float*    s_red    = (float*)(smem + SMEM_RED);
  float*    s_stat   = (float*)(smem + SMEM_STAT);
  float*    s_ctx    = (float*)(smem + SMEM_CTX);

  const int tid  = threadIdx.x;
  const int wave = __builtin_amdgcn_readfirstlane(tid >> 5);  // wave-uniform SGPR
  const int lane = tid & 31;
  const int ln16 = lane & 15;
  const int hiSel = (lane >= 16) ? 8 : 0;
  const int mBase = (lane >> 4) << 3;   // 0 or 8 (C/D fragment row base)

  const int qt = blockIdx.x & 63;                 // query tile (S/16 = 64)
  const int h  = (blockIdx.x >> 6) & (HEADS - 1);
  const int b  = blockIdx.x >> 10;
  const int qb = qt * 16;
  const size_t bh = (size_t)(b * HEADS + h);

  // ---- pass 1: scores S[16][1024] = Q(16x64) @ K^T, via WMMA f16 ----
  const _Float16* qrow = Qh + (bh * SEQ + qb + ln16) * DH;
  v16h a0 = load_frag16(qrow, hiSel);        // dh 0..31
  v16h a1 = load_frag16(qrow + 32, hiSel);   // dh 32..63

  // fixed trip count: uniform control flow, no exec-mask loop around WMMA
#pragma unroll 2
  for (int t = 0; t < (SEQ / 16) / 4; ++t) {
    const int kv0 = (wave + 4 * t) * 16;
    const _Float16* krow = Kh + (bh * SEQ + kv0 + ln16) * DH;
    __builtin_prefetch(krow + 64 * DH, 0, 1);  // next tile for this wave
    v16h b0 = load_frag16(krow, hiSel);
    v16h b1 = load_frag16(krow + 32, hiSel);
    v8f c = {};
    c = wmma_f16(a0, b0, c);
    c = wmma_f16(a1, b1, c);
#pragma unroll
    for (int i = 0; i < 8; ++i)
      s_scores[(mBase + i) * SEQ + kv0 + ln16] = c[i];
  }
  __syncthreads();

  // ---- softmax over 16 rows (8 threads/row, 128 contiguous cols each, f32x4) ----
  const int row = tid >> 3;
  const int sub = tid & 7;
  const int c0  = sub * 128;
  float* srow = s_scores + row * SEQ;
  v4f* srow4 = (v4f*)(srow + c0);

  float m = -3.402823466e38f;
#pragma unroll 8
  for (int j = 0; j < 32; ++j) {
    v4f x = srow4[j];
    m = fmaxf(m, fmaxf(fmaxf(x[0], x[1]), fmaxf(x[2], x[3])));
  }
  s_red[row * 8 + sub] = m;
  __syncthreads();
  if (sub == 0) {
    float mm = s_red[row * 8];
#pragma unroll
    for (int j = 1; j < 8; ++j) mm = fmaxf(mm, s_red[row * 8 + j]);
    s_stat[row] = mm;
  }
  __syncthreads();
  const float rmax = s_stat[row];

  float sum = 0.0f;
#pragma unroll 8
  for (int j = 0; j < 32; ++j) {
    v4f x = srow4[j];
    v4f e;
#pragma unroll
    for (int i = 0; i < 4; ++i) e[i] = __expf(x[i] - rmax);
    srow4[j] = e;
    sum += (e[0] + e[1]) + (e[2] + e[3]);
  }
  s_red[row * 8 + sub] = sum;
  __syncthreads();
  if (sub == 0) {
    float ss = s_red[row * 8];
#pragma unroll
    for (int j = 1; j < 8; ++j) ss += s_red[row * 8 + j];
    s_stat[row] = 1.0f / ss;
  }
  __syncthreads();
  const float inv = s_stat[row];

  float* attnRow = attn + (bh * SEQ + qb + row) * SEQ;
  v4f* arow4 = (v4f*)(attnRow + c0);
  v4h* prow4 = (v4h*)(s_p + row * SEQ + c0);
#pragma unroll 8
  for (int j = 0; j < 32; ++j) {
    v4f p = srow4[j];
    p *= inv;
    arow4[j] = p;                        // coalesced 16B attn stores (537 MB stream)
    v4h ph;
#pragma unroll
    for (int i = 0; i < 4; ++i) ph[i] = (_Float16)p[i];
    prow4[j] = ph;                       // f16 copy feeds the P@V WMMA pass
  }
  __syncthreads();

  // ---- pass 2: ctx[16][64] = P(16x1024) @ V(1024x64); one 16-col tile/wave ----
  const int col = wave * 16 + ln16;      // dh channel handled by this lane
  const _Float16* vb = Vt + (bh * DH + col) * SEQ;  // keys contiguous
  const _Float16* pb = s_p + ln16 * SEQ;            // A rows from LDS
  v8f cc = {};
#pragma unroll 4
  for (int kk = 0; kk < SEQ; kk += 32) {
    v16h af = load_frag16(pb + kk, hiSel);          // ds_load path
    v16h bf = load_frag16(vb + kk, hiSel);
    cc = wmma_f16(af, bf, cc);
  }
#pragma unroll
  for (int i = 0; i < 8; ++i)
    s_ctx[(mBase + i) * DH + col] = cc[i];
  __syncthreads();

  // cooperative coalesced f16 store of the 16x64 ctx tile (concat layout [b,s,D])
  {
    int idx = tid * 8;                   // 1024 elements / 128 threads
    int r = idx >> 6, c = idx & 63;
    v8h o;
#pragma unroll
    for (int i = 0; i < 8; ++i) o[i] = (_Float16)s_ctx[r * DH + c + i];
    *(v8h*)(Ctx + ((size_t)b * SEQ + qb + r) * DMODEL + h * DH + c) = o;
  }
}

// ---------------- kernel 4: out = Ctx @ W^T + bias (M=8192,N=1024,K=1024) ----
__global__ void __launch_bounds__(128)
dense_kernel(const _Float16* __restrict__ Ctx, const _Float16* __restrict__ Wh,
             const float* __restrict__ bias, float* __restrict__ out) {
  const int wave = __builtin_amdgcn_readfirstlane(threadIdx.x >> 5);
  const int lane = threadIdx.x & 31;
  const int ln16 = lane & 15;
  const int hiSel = (lane >= 16) ? 8 : 0;
  const int mBase = (lane >> 4) << 3;

  const int rowTile  = blockIdx.x >> 4;          // 0..511 (8192 rows / 16)
  const int colGroup = blockIdx.x & 15;          // 0..15  (1024 cols / 64)
  const int m0 = rowTile * 16;
  const int n  = colGroup * 64 + wave * 16 + ln16;

  const _Float16* arow = Ctx + (size_t)(m0 + ln16) * DMODEL;  // k contiguous
  const _Float16* brow = Wh + (size_t)n * DMODEL;             // W[n][k], k contiguous
  v8f c = {};
#pragma unroll 4
  for (int kk = 0; kk < DMODEL; kk += 32) {
    __builtin_prefetch(brow + kk + 256, 0, 1);
    v16h af = load_frag16(arow + kk, hiSel);
    v16h bf = load_frag16(brow + kk, hiSel);
    c = wmma_f16(af, bf, c);
  }
  const float bv = bias[n];
#pragma unroll
  for (int i = 0; i < 8; ++i)
    out[(size_t)(m0 + mBase + i) * DMODEL + n] = c[i] + bv;
}

// ---------------- launcher ----------------
extern "C" void kernel_launch(void* const* d_in, const int* in_sizes, int n_in,
                              void* d_out, int out_size, void* d_ws, size_t ws_size,
                              hipStream_t stream) {
  (void)in_sizes; (void)n_in; (void)out_size; (void)ws_size;
  const float* q  = (const float*)d_in[0];
  const float* k  = (const float*)d_in[1];
  const float* v  = (const float*)d_in[2];
  const float* w  = (const float*)d_in[3];
  const float* bb = (const float*)d_in[4];

  // workspace carve-up (all offsets 256B aligned)
  const size_t QKV_BYTES = (size_t)BATCH * SEQ * DMODEL * sizeof(_Float16); // 16 MB
  char* ws = (char*)d_ws;
  _Float16* Qh  = (_Float16*)(ws + 0 * QKV_BYTES);
  _Float16* Kh  = (_Float16*)(ws + 1 * QKV_BYTES);
  _Float16* Vt  = (_Float16*)(ws + 2 * QKV_BYTES);
  _Float16* Ctx = (_Float16*)(ws + 3 * QKV_BYTES);
  _Float16* Wh  = (_Float16*)(ws + 4 * QKV_BYTES);  // 2 MB

  float* out  = (float*)d_out;
  float* attn = out + (size_t)BATCH * SEQ * DMODEL;  // tuple output #2

  const size_t nElem = (size_t)BATCH * SEQ * DMODEL;
  convert_qkv<<<(unsigned)((nElem + 255) / 256), 256, 0, stream>>>(q, k, v, Qh, Kh, Vt);
  convert_w<<<(DMODEL * DMODEL + 255) / 256, 256, 0, stream>>>(w, Wh);

  // one block per (b, h, 16-row query tile): 8*16*64 = 8192 blocks, 4 waves each
  attn_kernel<<<BATCH * HEADS * (SEQ / 16), 128, SMEM_TOTAL, stream>>>(
      Qh, Kh, Vt, attn, Ctx);

  // (8192/16 row tiles) * (1024/64 col groups) = 8192 blocks
  dense_kernel<<<(BATCH * SEQ / 16) * (DMODEL / 64), 128, 0, stream>>>(
      Ctx, Wh, bb, out);
}
